// Seq2SeqLSTM_28475633173165
// MI455X (gfx1250) — matrix operand order repaired and checked by reading, not compiled, not run on hardware
//
#include <hip/hip_runtime.h>
#include <hip/hip_bf16.h>
#include <cmath>

// ---------------------------------------------------------------------------
// CDNA5 (gfx1250) wave32 WMMA types & fragment helpers
// ---------------------------------------------------------------------------
typedef __attribute__((ext_vector_type(16))) _Float16 v16h;
typedef __attribute__((ext_vector_type(8)))  _Float16 v8h;
typedef __attribute__((ext_vector_type(8)))  float    v8f;
typedef int vi4 __attribute__((vector_size(16)));   // matches builtin param type

#define DEV __device__ __forceinline__

// A fragment: 16x32 f16 tile, row-major source (lda elements per row).
// lanes 0-15: M=lane, K={0..7,16..23}; lanes 16-31: M=lane-16, K={8..15,24..31}.
DEV v16h load_a_frag(const _Float16* __restrict__ A, int lda, int m0, int k0, int lane) {
  int m  = m0 + (lane & 15);
  int kb = k0 + ((lane >> 4) << 3);            // 0 or 8
  const _Float16* p = A + m * lda + kb;
  v8h lo = *reinterpret_cast<const v8h*>(p);       // K = kb .. kb+7
  v8h hi = *reinterpret_cast<const v8h*>(p + 16);  // K = kb+16 .. kb+23
  return __builtin_shufflevector(lo, hi, 0,1,2,3,4,5,6,7,8,9,10,11,12,13,14,15);
}

// B fragment: 32x16 f16 tile = columns n0..n0+15 of W[N][K] (row-major, ldb).
DEV v16h load_b_frag(const _Float16* __restrict__ W, int ldb, int n0, int k0, int lane) {
  int n  = n0 + (lane & 15);
  int kb = k0 + ((lane >> 4) << 3);            // 0 or 8
  const _Float16* p = W + n * ldb + kb;
  v8h lo = *reinterpret_cast<const v8h*>(p);
  v8h hi = *reinterpret_cast<const v8h*>(p + 16);
  return __builtin_shufflevector(lo, hi, 0,1,2,3,4,5,6,7,8,9,10,11,12,13,14,15);
}

template <bool RA, bool RB>
DEV v8f wmma16(v16h a, v16h b, v8f c) {
  // (neg_a, A, neg_b, B, c_mod, C, reuse_a, reuse_b) -- modifiers are imms
  return __builtin_amdgcn_wmma_f32_16x16x32_f16(false, a, false, b, (short)0, c,
                                                RA, RB);
}

DEV float sigm(float x) { return 1.0f / (1.0f + __expf(-x)); }

// ---------------------------------------------------------------------------
// Async global->LDS staging (CDNA5 ASYNCcnt path) with safe fallback.
// ---------------------------------------------------------------------------
DEV void async_cp16(const float* __restrict__ g, float* __restrict__ l) {
#if __has_builtin(__builtin_amdgcn_global_load_async_to_lds_b128)
  __builtin_amdgcn_global_load_async_to_lds_b128(
      (__attribute__((address_space(1))) vi4*)const_cast<float*>(g),
      (__attribute__((address_space(3))) vi4*)l,
      0, 0);
#else
  *reinterpret_cast<float4*>(l) = *reinterpret_cast<const float4*>(g);
#endif
}
DEV void wait_async_le4() { asm volatile("s_wait_asynccnt 0x4" ::: "memory"); }
DEV void wait_async_0()   { asm volatile("s_wait_asynccnt 0x0" ::: "memory"); }

// ---------------------------------------------------------------------------
// One GEMM phase of the LSTM step: acc[mt*4+gate] += A[mt-tile] @ Wgate^T.
// One wave owns BOTH batch tiles (M=32 = 2x16): every B fragment is loaded
// once and used twice. Modulo-2 software pipeline: two named fragment sets
// (slice A = k0, slice B = k0+32); each set is refilled IN PLACE right after
// its WMMAs consume it, so there are no cur<-next register copies (and none
// of the WMMA->VALU hazard nops they force). Requires K % 64 == 0.
// ---------------------------------------------------------------------------
DEV void lstm_gemm_phase(const _Float16* __restrict__ A, int lda,
                         const _Float16* __restrict__ W, int K,
                         int n0, int lane, v8f* __restrict__ acc) {
  v16h a0A = load_a_frag(A, lda,  0, 0, lane);
  v16h a1A = load_a_frag(A, lda, 16, 0, lane);
  v16h b0A = load_b_frag(W, K, 0 * 512 + n0, 0, lane);
  v16h b1A = load_b_frag(W, K, 1 * 512 + n0, 0, lane);
  v16h b2A = load_b_frag(W, K, 2 * 512 + n0, 0, lane);
  v16h b3A = load_b_frag(W, K, 3 * 512 + n0, 0, lane);
  v16h a0B = load_a_frag(A, lda,  0, 32, lane);
  v16h a1B = load_a_frag(A, lda, 16, 32, lane);
  v16h b0B = load_b_frag(W, K, 0 * 512 + n0, 32, lane);
  v16h b1B = load_b_frag(W, K, 1 * 512 + n0, 32, lane);
  v16h b2B = load_b_frag(W, K, 2 * 512 + n0, 32, lane);
  v16h b3B = load_b_frag(W, K, 3 * 512 + n0, 32, lane);
#pragma unroll 1
  for (int k0 = 0; k0 < K; k0 += 64) {
    const int kn = k0 + 64;
    // ---- slice A (k0) : grouped by A operand for the reuse hint ----
    acc[0] = wmma16<false, false>(a0A, b0A, acc[0]);
    acc[1] = wmma16<true,  false>(a0A, b1A, acc[1]);
    acc[2] = wmma16<true,  false>(a0A, b2A, acc[2]);
    acc[3] = wmma16<true,  false>(a0A, b3A, acc[3]);
    acc[4] = wmma16<false, false>(a1A, b0A, acc[4]);
    acc[5] = wmma16<true,  false>(a1A, b1A, acc[5]);
    acc[6] = wmma16<true,  false>(a1A, b2A, acc[6]);
    acc[7] = wmma16<true,  false>(a1A, b3A, acc[7]);
    if (kn < K) {                       // refill slice A for k0+64
      a0A = load_a_frag(A, lda,  0, kn, lane);
      a1A = load_a_frag(A, lda, 16, kn, lane);
      b0A = load_b_frag(W, K, 0 * 512 + n0, kn, lane);
      b1A = load_b_frag(W, K, 1 * 512 + n0, kn, lane);
      b2A = load_b_frag(W, K, 2 * 512 + n0, kn, lane);
      b3A = load_b_frag(W, K, 3 * 512 + n0, kn, lane);
    }
    // ---- slice B (k0+32) ----
    acc[0] = wmma16<false, false>(a0B, b0B, acc[0]);
    acc[1] = wmma16<true,  false>(a0B, b1B, acc[1]);
    acc[2] = wmma16<true,  false>(a0B, b2B, acc[2]);
    acc[3] = wmma16<true,  false>(a0B, b3B, acc[3]);
    acc[4] = wmma16<false, false>(a1B, b0B, acc[4]);
    acc[5] = wmma16<true,  false>(a1B, b1B, acc[5]);
    acc[6] = wmma16<true,  false>(a1B, b2B, acc[6]);
    acc[7] = wmma16<true,  false>(a1B, b3B, acc[7]);
    if (kn < K) {                       // refill slice B for k0+96
      a0B = load_a_frag(A, lda,  0, kn + 32, lane);
      a1B = load_a_frag(A, lda, 16, kn + 32, lane);
      b0B = load_b_frag(W, K, 0 * 512 + n0, kn + 32, lane);
      b1B = load_b_frag(W, K, 1 * 512 + n0, kn + 32, lane);
      b2B = load_b_frag(W, K, 2 * 512 + n0, kn + 32, lane);
      b3B = load_b_frag(W, K, 3 * 512 + n0, kn + 32, lane);
    }
  }
}

// ---------------------------------------------------------------------------
// Fused LSTM step: one wave per 16 hidden units, all 4 gates & both batch
// tiles in-register. Grid: 8 blocks x 128 thr (4 waves) = 32 waves total, so
// occupancy is irrelevant -> __launch_bounds__(128,1) lifts the VGPR cap.
// ---------------------------------------------------------------------------
__global__ void __launch_bounds__(128, 1)
lstm_step_kernel(const _Float16* __restrict__ X, int K1,
                 const _Float16* __restrict__ Wih,
                 const _Float16* __restrict__ Whh,
                 const _Float16* __restrict__ Hprev,
                 const float* __restrict__ bias,
                 float* __restrict__ Cst,
                 _Float16* __restrict__ Hnew,
                 _Float16* __restrict__ Hcopy16,
                 float* __restrict__ Hcopy32) {
  const int lane = threadIdx.x & 31;
  const int wid  = blockIdx.x * (blockDim.x >> 5) + (threadIdx.x >> 5);
  const int n0   = wid << 4;           // hidden-unit tile: 0..496

  v8f acc[8] = {};
  lstm_gemm_phase(X,     K1,  Wih, K1,  n0, lane, acc);
  lstm_gemm_phase(Hprev, 512, Whh, 512, n0, lane, acc);

  const int n = n0 + (lane & 15);
  const float bi = bias[n];
  const float bf = bias[512 + n];
  const float bg = bias[1024 + n];
  const float bo = bias[1536 + n];
#pragma unroll
  for (int mt = 0; mt < 2; ++mt) {
    const int mb = (mt << 4) + ((lane >> 4) << 3);
    const v8f* G = acc + mt * 4;
#pragma unroll
    for (int r = 0; r < 8; ++r) {
      const int idx = (mb + r) * 512 + n;
      float gi = sigm(G[0][r] + bi);
      float gf = sigm(G[1][r] + bf);
      float gg = tanhf(G[2][r] + bg);
      float go = sigm(G[3][r] + bo);
      float c  = gf * Cst[idx] + gi * gg;
      float h  = go * tanhf(c);
      Cst[idx]  = c;                      // in-place safe: 1 reader==1 writer
      Hnew[idx] = (_Float16)h;
      if (Hcopy16) Hcopy16[idx] = (_Float16)h;
      if (Hcopy32) Hcopy32[idx] = h;
    }
  }
}

// ---------------------------------------------------------------------------
// Generic WMMA GEMM: C[M,N] = A[M,K] @ W[N,K]^T + bias[N].
// Same modulo-2 pipeline (K % 64 == 0 for all uses: K = 512).
// MAP==0: C[m*ldc+n].  MAP==1: logits store row m = t*32+b -> [B,T,V].
// ---------------------------------------------------------------------------
template <int MAP>
__global__ void wmma_gemm_kernel(const _Float16* __restrict__ A,
                                 const _Float16* __restrict__ W,
                                 const float* __restrict__ bias,
                                 float* __restrict__ C,
                                 int M, int N, int K, int ldc) {
  const int lane   = threadIdx.x & 31;
  const int wid    = blockIdx.x * (blockDim.x >> 5) + (threadIdx.x >> 5);
  const int nwaves = gridDim.x * (blockDim.x >> 5);
  const int mt = M >> 4, nt = N >> 4;
  for (int t = wid; t < mt * nt; t += nwaves) {
    const int m0 = (t / nt) << 4;
    const int n0 = (t % nt) << 4;
    v8f acc = {};
    v16h aA = load_a_frag(A, K, m0, 0, lane);
    v16h bA = load_b_frag(W, K, n0, 0, lane);
    v16h aB = load_a_frag(A, K, m0, 32, lane);
    v16h bB = load_b_frag(W, K, n0, 32, lane);
#pragma unroll 1
    for (int k0 = 0; k0 < K; k0 += 64) {
      const int kn = k0 + 64;
      acc = wmma16<false, false>(aA, bA, acc);
      if (kn < K) {
        aA = load_a_frag(A, K, m0, kn, lane);
        bA = load_b_frag(W, K, n0, kn, lane);
      }
      acc = wmma16<false, false>(aB, bB, acc);
      if (kn < K) {
        aB = load_a_frag(A, K, m0, kn + 32, lane);
        bB = load_b_frag(W, K, n0, kn + 32, lane);
      }
    }
    const int n  = n0 + (lane & 15);
    const int mb = m0 + ((lane >> 4) << 3);
    const float bn2 = bias ? bias[n] : 0.0f;
#pragma unroll
    for (int r = 0; r < 8; ++r) {
      const int m = mb + r;
      const float v = acc[r] + bn2;
      if (MAP == 0) {
        C[m * ldc + n] = v;
      } else {
        C[((m & 31) * 256 + (m >> 5)) * 128 + n] = v;  // [B,T,V]
      }
    }
  }
}

// ---------------------------------------------------------------------------
// Attention scores: scores[b,s] = Va_b + sum_h Va[h]*tanh(qW[b,h]+Uk[s*32+b,h])
// ---------------------------------------------------------------------------
__global__ void attn_score_kernel(const float* __restrict__ qW,
                                  const float* __restrict__ Uk,
                                  const float* __restrict__ Va,
                                  const float* __restrict__ Vab,
                                  float* __restrict__ scores) {
  const int wid  = blockIdx.x * (blockDim.x >> 5) + (threadIdx.x >> 5);
  const int lane = threadIdx.x & 31;
  if (wid >= 32 * 256) return;
  const int b = wid & 31;
  const float* uk = Uk + (size_t)wid * 512;
  const float* qw = qW + b * 512;
  float sum = 0.0f;
  for (int h = lane; h < 512; h += 32) sum += Va[h] * tanhf(qw[h] + uk[h]);
  for (int o = 16; o > 0; o >>= 1) sum += __shfl_xor(sum, o, 32);
  if (lane == 0) scores[b * 256 + (wid >> 5)] = sum + Vab[0];
}

// ---------------------------------------------------------------------------
// Softmax over S + context + decoder-input assembly. One block per batch row.
// keys rows are staged through LDS with double-buffered async global->LDS
// copies: issue chunk c+1, s_wait_asynccnt<=4 (the 4 in-flight ops of chunk c
// are the oldest, ASYNCcnt completes in order), barrier, consume chunk c.
// ---------------------------------------------------------------------------
__global__ void softmax_ctx_kernel(const float* __restrict__ scores,
                                   const float* __restrict__ keys32,
                                   const _Float16* __restrict__ tgt_t,
                                   _Float16* __restrict__ xcat) {
  __shared__ float red[256];
  __shared__ float sw[256];
  __shared__ float kt[2][8 * 512];          // 2 x 16KB chunk buffers
  const int b = blockIdx.x, tid = threadIdx.x;

  // softmax over S=256 scores
  const float v = scores[b * 256 + tid];
  red[tid] = v; __syncthreads();
  for (int o = 128; o > 0; o >>= 1) {
    if (tid < o) red[tid] = fmaxf(red[tid], red[tid + o]);
    __syncthreads();
  }
  const float mx = red[0]; __syncthreads();
  const float e = __expf(v - mx);
  red[tid] = e; __syncthreads();
  for (int o = 128; o > 0; o >>= 1) {
    if (tid < o) red[tid] += red[tid + o];
    __syncthreads();
  }
  sw[tid] = e / red[0];

  // stage one 8-row chunk: 1024 float4s, 4 async ops per wave
  auto issue_chunk = [&](int c, float* dst) {
#pragma unroll
    for (int j = 0; j < 4; ++j) {
      const int idx = j * 256 + tid;        // float4 index within chunk
      const int row = idx >> 7;             // 0..7
      const int w4  = idx & 127;
      const float* src = keys32 + ((size_t)(c * 8 + row) * 32 + b) * 512 + w4 * 4;
      async_cp16(src, dst + row * 512 + w4 * 4);
    }
  };

  issue_chunk(0, kt[0]);
  float ctx0 = 0.0f, ctx1 = 0.0f;
  for (int c = 0; c < 32; ++c) {
    float* cur = kt[c & 1];
    if (c + 1 < 32) {
      issue_chunk(c + 1, kt[(c + 1) & 1]);
      wait_async_le4();                     // chunk c complete, c+1 in flight
    } else {
      wait_async_0();
    }
    __syncthreads();                        // all waves' copies visible
#pragma unroll
    for (int j = 0; j < 8; ++j) {
      const float w = sw[c * 8 + j];
      ctx0 += w * cur[j * 512 + tid];
      ctx1 += w * cur[j * 512 + tid + 256];
    }
    __syncthreads();                        // done reading before overwrite
  }
  xcat[b * 1024 + 512 + tid]       = (_Float16)ctx0;
  xcat[b * 1024 + 512 + tid + 256] = (_Float16)ctx1;
  xcat[b * 1024 + tid]             = tgt_t[b * 512 + tid];
  xcat[b * 1024 + tid + 256]       = tgt_t[b * 512 + tid + 256];
}

// ---------------------------------------------------------------------------
// Small utility kernels
// ---------------------------------------------------------------------------
__global__ void f32_to_f16_kernel(const float* __restrict__ in,
                                  _Float16* __restrict__ out, int n) {
  int i = blockIdx.x * blockDim.x + threadIdx.x;
  if (i < n) out[i] = (_Float16)in[i];
}
__global__ void add_f32_kernel(const float* __restrict__ a,
                               const float* __restrict__ b,
                               float* __restrict__ o, int n) {
  int i = blockIdx.x * blockDim.x + threadIdx.x;
  if (i < n) o[i] = a[i] + b[i];
}
__global__ void zero_f16_kernel(_Float16* p, int n) {
  int i = blockIdx.x * blockDim.x + threadIdx.x;
  if (i < n) p[i] = (_Float16)0.0f;
}
__global__ void zero_f32_kernel(float* p, int n) {
  int i = blockIdx.x * blockDim.x + threadIdx.x;
  if (i < n) p[i] = 0.0f;
}
// x:[B,S] int32 -> out:[S][B][H] f16 (row = s*32+b)
__global__ void embed_kernel(const int* __restrict__ idx,
                             const float* __restrict__ emb,
                             _Float16* __restrict__ out, int total) {
  int i = blockIdx.x * blockDim.x + threadIdx.x;
  if (i >= total) return;
  int h = i & 511;
  int rb = i >> 9;
  int b = rb & 31, s = rb >> 5;
  out[i] = (_Float16)emb[(size_t)idx[b * 256 + s] * 512 + h];
}

// ---------------------------------------------------------------------------
// Host orchestration
// ---------------------------------------------------------------------------
extern "C" void kernel_launch(void* const* d_in, const int* in_sizes, int n_in,
                              void* d_out, int out_size, void* d_ws, size_t ws_size,
                              hipStream_t stream) {
  (void)in_sizes; (void)n_in; (void)out_size; (void)ws_size;
  const int B = 32, S = 256, T = 256, H = 512;

  const int*   x        = (const int*)d_in[0];
  const int*   tgt      = (const int*)d_in[1];
  const float* emb      = (const float*)d_in[2];
  const float* encWih   = (const float*)d_in[3];
  const float* encWhh   = (const float*)d_in[4];
  const float* enc_bih  = (const float*)d_in[5];
  const float* enc_bhh  = (const float*)d_in[6];
  const float* decWih0  = (const float*)d_in[7];
  const float* decWih1  = (const float*)d_in[8];
  const float* decWhh   = (const float*)d_in[9];
  const float* dec_bih  = (const float*)d_in[10];
  const float* dec_bhh  = (const float*)d_in[11];
  const float* Wa_w     = (const float*)d_in[12];
  const float* Wa_b     = (const float*)d_in[13];
  const float* Ua_w     = (const float*)d_in[14];
  const float* Ua_b     = (const float*)d_in[15];
  const float* Va_w     = (const float*)d_in[16];
  const float* Va_b     = (const float*)d_in[17];
  const float* out_w    = (const float*)d_in[18];
  const float* out_b    = (const float*)d_in[19];

  char* base = (char*)d_ws;
  size_t off = 0;
  auto alloc  = [&](size_t bytes) -> void* {
    off = (off + 255) & ~size_t(255);
    void* p = base + off; off += bytes; return p;
  };
  auto allocH = [&](size_t n) -> _Float16* { return (_Float16*)alloc(n * 2); };
  auto allocF = [&](size_t n) -> float*    { return (float*)alloc(n * 4); };

  _Float16* encWih16  = allocH((size_t)2 * 2048 * 512);
  _Float16* encWhh16  = allocH((size_t)2 * 2048 * 512);
  _Float16* decWih0_16= allocH((size_t)2048 * 1024);
  _Float16* decWih1_16= allocH((size_t)2048 * 512);
  _Float16* decWhh16  = allocH((size_t)2 * 2048 * 512);
  _Float16* Wa16      = allocH((size_t)512 * 512);
  _Float16* Ua16      = allocH((size_t)512 * 512);
  _Float16* outW16    = allocH((size_t)128 * 512);
  float* encB = allocF(4096);
  float* decB = allocF(4096);
  _Float16* xemb     = allocH((size_t)S * B * H);
  _Float16* tgt16    = allocH((size_t)T * B * H);
  _Float16* keys16   = allocH((size_t)S * B * H);
  float*    keys32   = allocF((size_t)S * B * H);
  float*    Uk       = allocF((size_t)S * B * H);
  _Float16* decseq16 = allocH((size_t)T * B * H);
  float*    qW       = allocF((size_t)B * H);
  float*    scores   = allocF((size_t)B * S);
  _Float16* xcat     = allocH((size_t)B * 2 * H);
  _Float16* hb = allocH((size_t)8 * B * H);
  float*    cb = allocF((size_t)4 * B * H);
  _Float16* eH0[2] = { hb + 0 * B * H, hb + 1 * B * H };
  _Float16* eH1[2] = { hb + 2 * B * H, hb + 3 * B * H };
  _Float16* dH0[2] = { hb + 4 * B * H, hb + 5 * B * H };
  _Float16* dH1[2] = { hb + 6 * B * H, hb + 7 * B * H };
  float *eC0 = cb, *eC1 = cb + B * H, *dC0 = cb + 2 * B * H, *dC1 = cb + 3 * B * H;

  auto cvt = [&](const float* src, _Float16* dst, int n) {
    f32_to_f16_kernel<<<(n + 255) / 256, 256, 0, stream>>>(src, dst, n);
  };

  // ---- prep ----
  cvt(encWih,  encWih16,   2 * 2048 * 512);
  cvt(encWhh,  encWhh16,   2 * 2048 * 512);
  cvt(decWih0, decWih0_16, 2048 * 1024);
  cvt(decWih1, decWih1_16, 2048 * 512);
  cvt(decWhh,  decWhh16,   2 * 2048 * 512);
  cvt(Wa_w, Wa16, 512 * 512);
  cvt(Ua_w, Ua16, 512 * 512);
  cvt(out_w, outW16, 128 * 512);
  add_f32_kernel<<<16, 256, 0, stream>>>(enc_bih, enc_bhh, encB, 4096);
  add_f32_kernel<<<16, 256, 0, stream>>>(dec_bih, dec_bhh, decB, 4096);
  zero_f16_kernel<<<(8 * B * H + 255) / 256, 256, 0, stream>>>(hb, 8 * B * H);
  zero_f32_kernel<<<(4 * B * H + 255) / 256, 256, 0, stream>>>(cb, 4 * B * H);
  embed_kernel<<<(S * B * H + 255) / 256, 256, 0, stream>>>(x,   emb, xemb,  S * B * H);
  embed_kernel<<<(T * B * H + 255) / 256, 256, 0, stream>>>(tgt, emb, tgt16, T * B * H);

  // ---- encoder ----
  for (int s = 0; s < S; ++s) {
    const int p = s & 1, q = p ^ 1;
    lstm_step_kernel<<<8, 128, 0, stream>>>(
        xemb + (size_t)s * B * H, H, encWih16, encWhh16,
        eH0[p], encB, eC0, eH0[q], nullptr, nullptr);
    lstm_step_kernel<<<8, 128, 0, stream>>>(
        eH0[q], H, encWih16 + (size_t)2048 * 512, encWhh16 + (size_t)2048 * 512,
        eH1[p], encB + 2048, eC1, eH1[q],
        keys16 + (size_t)s * B * H, keys32 + (size_t)s * B * H);
  }

  // ---- Uk = keys @ Ua^T + Ua_b ----
  wmma_gemm_kernel<0><<<256, 256, 0, stream>>>(keys16, Ua16, Ua_b, Uk,
                                               S * B, 512, 512, 512);

  // ---- decoder ----
  for (int t = 0; t < T; ++t) {
    const int p = t & 1, q = p ^ 1;
    wmma_gemm_kernel<0><<<8, 256, 0, stream>>>(dH1[p], Wa16, Wa_b, qW,
                                               32, 512, 512, 512);
    attn_score_kernel<<<1024, 256, 0, stream>>>(qW, Uk, Va_w, Va_b, scores);
    softmax_ctx_kernel<<<32, 256, 0, stream>>>(scores, keys32,
                                               tgt16 + (size_t)t * B * H, xcat);
    lstm_step_kernel<<<8, 128, 0, stream>>>(
        xcat, 2 * H, decWih0_16, decWhh16,
        dH0[p], decB, dC0, dH0[q], nullptr, nullptr);
    lstm_step_kernel<<<8, 128, 0, stream>>>(
        dH0[q], H, decWih1_16, decWhh16 + (size_t)2048 * 512,
        dH1[p], decB + 2048, dC1, dH1[q],
        decseq16 + (size_t)t * B * H, nullptr);
  }

  // ---- logits ----
  wmma_gemm_kernel<1><<<128, 256, 0, stream>>>(decseq16, outW16, out_b,
                                               (float*)d_out, T * B, 128, 512, 0);
}
